// Block_90305982365890
// MI455X (gfx1250) — matrix operand order repaired and checked by reading, not compiled, not run on hardware
//
#include <hip/hip_runtime.h>
#include <stdint.h>

// ---------------------------------------------------------------------------
// Transformer block for MI455X (gfx1250): bf16 WMMA GEMMs + flash attention,
// async global->LDS staging (ASYNCcnt), double-buffered GEMM tiles,
// transposed V staging so all WMMA fragments load contiguous from LDS.
// B=1, T=2048, D=1024, H=16, HS=64, FF=4096.
// ---------------------------------------------------------------------------

#define T_SEQ 2048
#define DMODEL 1024
#define NHEAD 16
#define HSZ 64
#define FFDIM 4096

typedef __bf16  v16bf __attribute__((ext_vector_type(16)));
typedef float   v8f   __attribute__((ext_vector_type(8)));

union Frag {
    v16bf v;
    unsigned short u[16];
};

union U4 {
    uint4 q;
    unsigned short s[8];
};

__device__ __forceinline__ unsigned short f2bf(float f) {
    unsigned u = __builtin_bit_cast(unsigned, f);
    u += 0x7fffu + ((u >> 16) & 1u);      // round-to-nearest-even
    return (unsigned short)(u >> 16);
}

// LDS byte offset of a __shared__ pointer (generic ptr low 32 bits = LDS addr)
__device__ __forceinline__ unsigned lds_off(const void* p) {
    return (unsigned)(size_t)p;
}

// CDNA5 async global->LDS copy, 16 bytes per lane. Tracked by ASYNCcnt.
__device__ __forceinline__ void async_copy_b128(unsigned lds_addr, const void* gaddr) {
    asm volatile("global_load_async_to_lds_b128 %0, %1, off"
                 :: "v"(lds_addr), "v"(gaddr)
                 : "memory");
}

__device__ __forceinline__ void wait_async0() {
    asm volatile("s_wait_asynccnt 0" ::: "memory");
}

// ---------------------------------------------------------------------------
// Batched tiled transpose + fp32->bf16: in[z][R][C] -> out[z][C][R].
// 32x32 tiles through LDS; both global sides coalesced. R,C multiples of 32.
// ---------------------------------------------------------------------------
__global__ __launch_bounds__(256) void k_transpose2d_b(
    const float* __restrict__ in, unsigned short* __restrict__ out, int R, int C) {
    __shared__ unsigned short tile[32][33];
    const float* inb = in + (size_t)blockIdx.z * R * C;
    unsigned short* outb = out + (size_t)blockIdx.z * R * C;
    const int tr = blockIdx.x * 32;      // row tile origin
    const int tc = blockIdx.y * 32;      // col tile origin
    const int lc = threadIdx.x & 31;
    const int lr = threadIdx.x >> 5;     // 0..7
#pragma unroll
    for (int j = 0; j < 4; ++j) {
        const int r = lr + j * 8;
        tile[r][lc] = f2bf(inb[(size_t)(tr + r) * C + tc + lc]);
    }
    __syncthreads();
#pragma unroll
    for (int j = 0; j < 4; ++j) {
        const int c = lr + j * 8;
        outb[(size_t)(tc + c) * R + tr + lc] = tile[lc][c];
    }
}

// ---------------------------------------------------------------------------
// RMSNorm: out_bf16[t, :] = (x[t,:] * rsqrt(mean(x^2)+eps)) * g
// ---------------------------------------------------------------------------
__global__ __launch_bounds__(256) void k_rmsnorm(const float* __restrict__ x,
                                                 const float* __restrict__ g,
                                                 unsigned short* __restrict__ out) {
    __shared__ float part[8];
    const int t = blockIdx.x;
    const int tid = threadIdx.x;
    const float* row = x + t * DMODEL;
    float v[4];
    float s = 0.f;
#pragma unroll
    for (int j = 0; j < 4; ++j) {
        v[j] = row[tid + j * 256];
        s += v[j] * v[j];
    }
#pragma unroll
    for (int m = 16; m >= 1; m >>= 1) s += __shfl_xor(s, m, 32);
    if ((tid & 31) == 0) part[tid >> 5] = s;
    __syncthreads();
    if (tid == 0) {
        float tot = 0.f;
#pragma unroll
        for (int k = 0; k < 8; ++k) tot += part[k];
        part[0] = rsqrtf(tot * (1.0f / DMODEL) + 1e-6f);
    }
    __syncthreads();
    const float rn = part[0];
#pragma unroll
    for (int j = 0; j < 4; ++j)
        out[t * DMODEL + tid + j * 256] = f2bf(v[j] * rn * g[tid + j * 256]);
}

// ---------------------------------------------------------------------------
// WMMA GEMM: C[M,N] = A[M,K](bf16) * B[N,K](bf16)^T  (+bias)(+residual)(SiLU)
// Block: 256 threads = 8 waves; 64x128 C tile; wave = 16 rows x 64 cols
// (4x v_wmma_f32_16x16x32_bf16 per K step, A fragment reused 4x).
// Double-buffered LDS fed by async global->LDS; WMMA of tile k overlaps the
// async fetch of tile k+1.  M%64==0, N%128==0, K%32==0 (true for all shapes).
// ---------------------------------------------------------------------------
template <bool OUT_BF16, bool SILU>
__global__ __launch_bounds__(256) void k_gemm_bt(
    const unsigned short* __restrict__ A, int lda,
    const unsigned short* __restrict__ B, int ldb,
    float* __restrict__ outF, unsigned short* __restrict__ outB, int ldc,
    const float* __restrict__ bias, const float* __restrict__ residual,
    int K) {
    __shared__ unsigned short sA[2][64 * 32];    // 2 x 4KB
    __shared__ unsigned short sB[2][128 * 32];   // 2 x 8KB

    const int tid = threadIdx.x;
    const int mBase = blockIdx.x * 64;
    const int nBase = blockIdx.y * 128;
    const int wave = tid >> 5, lane = tid & 31;
    const int laneLo = lane & 15, hi = lane >> 4;
    const int wm = wave & 3;        // 16-row subtile
    const int wn = wave >> 2;       // 64-col half

    v8f acc[4] = {};

    // cooperative async staging: A 1x16B/thread, B 2x16B/thread
    const int alr = tid >> 2, alc = (tid & 3) * 8;
    const int blr = tid >> 1, blc = (tid & 1) * 16;
    const unsigned short* gA = A + (mBase + alr) * lda + alc;
    const unsigned short* gB = B + (nBase + blr) * ldb + blc;
    const unsigned ldsA0 = lds_off(&sA[0][alr * 32 + alc]);
    const unsigned ldsA1 = lds_off(&sA[1][alr * 32 + alc]);
    const unsigned ldsB0 = lds_off(&sB[0][blr * 32 + blc]);
    const unsigned ldsB1 = lds_off(&sB[1][blr * 32 + blc]);

    // prologue: fetch tile 0
    async_copy_b128(ldsA0, gA);
    async_copy_b128(ldsB0, gB);
    async_copy_b128(ldsB0 + 16, gB + 8);
    wait_async0();
    __syncthreads();

    int cur = 0;
    for (int k0 = 0; k0 < K; k0 += 32) {
        // kick off next tile into the other buffer (overlaps WMMA below)
        if (k0 + 32 < K) {
            const unsigned la = cur ? ldsA0 : ldsA1;
            const unsigned lb = cur ? ldsB0 : ldsB1;
            async_copy_b128(la, gA + k0 + 32);
            async_copy_b128(lb, gB + k0 + 32);
            async_copy_b128(lb + 16, gB + k0 + 40);
        }

        // A fragment: lane half selects K {0..7,16..23} vs {8..15,24..31}
        Frag a;
        const int ar = (wm * 16 + laneLo) * 32 + hi * 8;
#pragma unroll
        for (int j = 0; j < 8; ++j) {
            a.u[j]     = sA[cur][ar + j];
            a.u[8 + j] = sA[cur][ar + 16 + j];
        }
#pragma unroll
        for (int s = 0; s < 4; ++s) {
            // B fragment: lane = column, K {0..7,16..23} vs {8..15,24..31} per lane-half
            Frag b;
            const int br = (wn * 64 + s * 16 + laneLo) * 32 + hi * 8;
#pragma unroll
            for (int j = 0; j < 16; ++j) b.u[j] = sB[cur][br + j + (j >= 8 ? 8 : 0)];
            acc[s] = __builtin_amdgcn_wmma_f32_16x16x32_bf16(
                false, a.v, false, b.v, (short)0, acc[s], false, false);
        }

        wait_async0();       // own async writes to the other buffer done
        __syncthreads();     // everyone's writes visible / reads finished
        cur ^= 1;
    }

    // epilogue: C layout -> VGPR r = row r (+8 upper half), lane&15 = column
    const int rowBase = mBase + wm * 16 + hi * 8;
    const int colBase = nBase + wn * 64 + laneLo;
#pragma unroll
    for (int r = 0; r < 8; ++r) {
        const int row = rowBase + r;
#pragma unroll
        for (int s = 0; s < 4; ++s) {
            const int col = colBase + s * 16;
            float f = acc[s][r];
            if (bias) f += bias[col];
            if (residual) f += residual[row * ldc + col];
            if (SILU) f = f / (1.f + __expf(-f));
            if (OUT_BF16) outB[row * ldc + col] = f2bf(f);
            else          outF[row * ldc + col] = f;
        }
    }
}

// ---------------------------------------------------------------------------
// Flash attention (causal), one 64-query block x one head per 128-thread block.
// qkv layout: [T, 3*D] bf16; q at col h*64, k at 1024+h*64, v at 2048+h*64.
// QK^T and PV via v_wmma_f32_16x16x32_bf16; online softmax with per-row stats
// in registers (row = (VGPR r, lane-half), reduced over 16 lanes by shuffles).
// Q/K tiles staged with async global->LDS; V staged *transposed* ([hs][kv])
// through registers so the PV B-fragments are contiguous (vector ds loads).
// ---------------------------------------------------------------------------
__global__ __launch_bounds__(128) void k_attn(const unsigned short* __restrict__ qkv,
                                              unsigned short* __restrict__ attn) {
    __shared__ unsigned short sQ[64 * 64];
    __shared__ unsigned short sK[64 * 64];
    __shared__ unsigned short sVT[64 * 64];  // V transposed: [hs][kv]
    __shared__ unsigned short sP[64 * 64];   // 16x64 per wave (wave-private)

    const int tid = threadIdx.x;
    const int qBase = blockIdx.x * 64;
    const int h = blockIdx.y;
    const int wave = tid >> 5, lane = tid & 31;
    const int laneLo = lane & 15, hi = lane >> 4;
    const float scale = 0.125f;              // 1/sqrt(64)

    const int sr = tid >> 1, shalf = tid & 1;      // staging: 32 ushorts/thread
    const unsigned ldsQ = lds_off(&sQ[sr * 64 + shalf * 32]);
    const unsigned ldsK = lds_off(&sK[sr * 64 + shalf * 32]);

    // ---- stage Q tile (async) ----
    {
        const unsigned short* src = qkv + (qBase + sr) * 3072 + h * 64 + shalf * 32;
#pragma unroll
        for (int c = 0; c < 4; ++c) async_copy_b128(ldsQ + c * 16, src + c * 8);
    }
    wait_async0();
    __syncthreads();

    Frag qa[2];
    {
        const int base = (wave * 16 + laneLo) * 64 + hi * 8;
#pragma unroll
        for (int ks = 0; ks < 2; ++ks)
#pragma unroll
            for (int j = 0; j < 8; ++j) {
                qa[ks].u[j]     = sQ[base + ks * 32 + j];
                qa[ks].u[8 + j] = sQ[base + ks * 32 + 16 + j];
            }
    }

    v8f Ov[4] = {};
    float mstat[8], lstat[8];
#pragma unroll
    for (int r = 0; r < 8; ++r) { mstat[r] = -3.0e38f; lstat[r] = 0.f; }

    for (int kb = 0; kb <= (int)blockIdx.x; ++kb) {
        const int kvBase = kb * 64;
        // ---- stage K (async) and V (registers, transposed into sVT) ----
        {
            const unsigned short* sk = qkv + (kvBase + sr) * 3072 + 1024 + h * 64 + shalf * 32;
#pragma unroll
            for (int c = 0; c < 4; ++c) async_copy_b128(ldsK + c * 16, sk + c * 8);

            const unsigned short* sv = qkv + (kvBase + sr) * 3072 + 2048 + h * 64 + shalf * 32;
            U4 t[4];
#pragma unroll
            for (int c = 0; c < 4; ++c) t[c].q = *(const uint4*)(sv + c * 8);
#pragma unroll
            for (int c = 0; c < 4; ++c)
#pragma unroll
                for (int e = 0; e < 8; ++e) {
                    const int hh = shalf * 32 + c * 8 + e;     // hs index
                    sVT[hh * 64 + sr] = t[c].s[e];             // [hs][kv]
                }
        }
        wait_async0();
        __syncthreads();

        // ---- S = Q K^T (16x64 per wave) ----
        v8f Sv[4] = {};
#pragma unroll
        for (int ks = 0; ks < 2; ++ks) {
#pragma unroll
            for (int j = 0; j < 4; ++j) {
                Frag b;
                const int n = j * 16 + laneLo;      // key index within tile
                const int base = n * 64 + ks * 32 + hi * 8;
#pragma unroll
                for (int jj = 0; jj < 16; ++jj) b.u[jj] = sK[base + jj + (jj >= 8 ? 8 : 0)];
                Sv[j] = __builtin_amdgcn_wmma_f32_16x16x32_bf16(
                    false, qa[ks].v, false, b.v, (short)0, Sv[j], false, false);
            }
        }

        // ---- causal mask + online softmax update ----
#pragma unroll
        for (int r = 0; r < 8; ++r) {
            const int t = qBase + wave * 16 + hi * 8 + r;
            float sv[4];
            float mloc = -3.0e38f;
#pragma unroll
            for (int j = 0; j < 4; ++j) {
                const int u = kvBase + j * 16 + laneLo;
                const float x = (u <= t) ? Sv[j][r] * scale : -3.0e38f;
                sv[j] = x;
                mloc = fmaxf(mloc, x);
            }
#pragma unroll
            for (int d = 1; d < 16; d <<= 1) mloc = fmaxf(mloc, __shfl_xor(mloc, d, 32));
            const float mnew = fmaxf(mstat[r], mloc);
            const float corr = __expf(mstat[r] - mnew);
            float rs = 0.f;
            const int prow = (wave * 16 + hi * 8 + r) * 64 + laneLo;
#pragma unroll
            for (int j = 0; j < 4; ++j) {
                const float p = __expf(sv[j] - mnew);
                rs += p;
                sP[prow + j * 16] = f2bf(p);
            }
#pragma unroll
            for (int d = 1; d < 16; d <<= 1) rs += __shfl_xor(rs, d, 32);
            lstat[r] = lstat[r] * corr + rs;
            mstat[r] = mnew;
#pragma unroll
            for (int j = 0; j < 4; ++j) Ov[j][r] *= corr;
        }

        // ---- O += P V  (A = P from wave-private LDS, B = V^T from sVT) ----
#pragma unroll
        for (int ks = 0; ks < 2; ++ks) {
            Frag pa;
            const int base = (wave * 16 + laneLo) * 64 + ks * 32 + hi * 8;
#pragma unroll
            for (int j = 0; j < 8; ++j) {
                pa.u[j]     = sP[base + j];
                pa.u[8 + j] = sP[base + 16 + j];
            }
#pragma unroll
            for (int j = 0; j < 4; ++j) {
                Frag vb;
                const int n = j * 16 + laneLo;              // hs index
                const int vbase = n * 64 + ks * 32 + hi * 8;
#pragma unroll
                for (int jj = 0; jj < 16; ++jj) vb.u[jj] = sVT[vbase + jj + (jj >= 8 ? 8 : 0)];
                Ov[j] = __builtin_amdgcn_wmma_f32_16x16x32_bf16(
                    false, pa.v, false, vb.v, (short)0, Ov[j], false, false);
            }
        }
        __syncthreads();   // protect sK/sVT before next iteration's staging
    }

    // ---- write O / l as bf16 into attn[T, H*HS] ----
#pragma unroll
    for (int r = 0; r < 8; ++r) {
        const int t = qBase + wave * 16 + hi * 8 + r;
        const float inv = 1.0f / lstat[r];
#pragma unroll
        for (int j = 0; j < 4; ++j)
            attn[t * DMODEL + h * 64 + j * 16 + laneLo] = f2bf(Ov[j][r] * inv);
    }
}

// ---------------------------------------------------------------------------
// Host-side orchestration
// ---------------------------------------------------------------------------
extern "C" void kernel_launch(void* const* d_in, const int* in_sizes, int n_in,
                              void* d_out, int out_size, void* d_ws, size_t ws_size,
                              hipStream_t stream) {
    const float* x     = (const float*)d_in[0];
    const float* Wq    = (const float*)d_in[1];
    const float* Wk    = (const float*)d_in[2];
    const float* Wv    = (const float*)d_in[3];
    const float* Wproj = (const float*)d_in[4];
    const float* bproj = (const float*)d_in[5];
    const float* W1    = (const float*)d_in[6];
    const float* W2    = (const float*)d_in[7];
    const float* g1    = (const float*)d_in[8];
    const float* g2    = (const float*)d_in[9];

    char* ws = (char*)d_ws;
    unsigned short* wqkvT  = (unsigned short*)(ws + 0);          // [3072,1024] bf16  (6 MB)
    unsigned short* wprojT = (unsigned short*)(ws + 6291456);    // [1024,1024] bf16  (2 MB)
    unsigned short* w1T    = (unsigned short*)(ws + 8388608);    // [4096,1024] bf16  (8 MB)
    unsigned short* w2T    = (unsigned short*)(ws + 16777216);   // [1024,4096] bf16  (8 MB)
    unsigned short* xn     = (unsigned short*)(ws + 25165824);   // [2048,1024] bf16  (4 MB)
    unsigned short* qkv    = (unsigned short*)(ws + 29360128);   // [2048,3072] bf16  (12 MB)
    unsigned short* attnb  = (unsigned short*)(ws + 41943040);   // [2048,1024] bf16  (4 MB)
    float*          x1     = (float*)        (ws + 46137344);   // [2048,1024] f32   (8 MB)
    unsigned short* xn2    = (unsigned short*)(ws + 54525952);   // [2048,1024] bf16  (4 MB)
    unsigned short* hbf    = (unsigned short*)(ws + 58720256);   // [2048,4096] bf16  (16 MB)

    // --- weight prep: coalesced tiled transposes, fp32 -> bf16 [N,K] ---
    k_transpose2d_b<<<dim3(32, 2, 16), 256, 0, stream>>>(Wq, wqkvT, 1024, 64);
    k_transpose2d_b<<<dim3(32, 2, 16), 256, 0, stream>>>(Wk, wqkvT + 1024 * 1024, 1024, 64);
    k_transpose2d_b<<<dim3(32, 2, 16), 256, 0, stream>>>(Wv, wqkvT + 2048 * 1024, 1024, 64);
    k_transpose2d_b<<<dim3(32, 32, 1), 256, 0, stream>>>(Wproj, wprojT, 1024, 1024);
    k_transpose2d_b<<<dim3(32, 128, 1), 256, 0, stream>>>(W1, w1T, 1024, 4096);
    k_transpose2d_b<<<dim3(128, 32, 1), 256, 0, stream>>>(W2, w2T, 4096, 1024);

    // --- xn = RMSNorm(x) * g1 ---
    k_rmsnorm<<<T_SEQ, 256, 0, stream>>>(x, g1, xn);

    // --- qkv = xn @ [Wq|Wk|Wv]   (M=2048, N=3072, K=1024) ---
    k_gemm_bt<true, false><<<dim3(32, 24), 256, 0, stream>>>(
        xn, DMODEL, wqkvT, DMODEL, nullptr, qkv, 3072, nullptr, nullptr, DMODEL);

    // --- causal flash attention ---
    k_attn<<<dim3(T_SEQ / 64, NHEAD), 128, 0, stream>>>(qkv, attnb);

    // --- x1 = x + attn @ Wproj + bproj   (M=2048, N=1024, K=1024) ---
    k_gemm_bt<false, false><<<dim3(32, 8), 256, 0, stream>>>(
        attnb, DMODEL, wprojT, DMODEL, x1, nullptr, DMODEL, bproj, x, DMODEL);

    // --- xn2 = RMSNorm(x1) * g2 ---
    k_rmsnorm<<<T_SEQ, 256, 0, stream>>>(x1, g2, xn2);

    // --- h = SiLU(xn2 @ W1)   (M=2048, N=4096, K=1024) ---
    k_gemm_bt<true, true><<<dim3(32, 32), 256, 0, stream>>>(
        xn2, DMODEL, w1T, DMODEL, nullptr, hbf, FFDIM, nullptr, nullptr, DMODEL);

    // --- out = x1 + h @ W2   (M=2048, N=1024, K=4096) ---
    k_gemm_bt<false, false><<<dim3(32, 8), 256, 0, stream>>>(
        hbf, FFDIM, w2T, FFDIM, (float*)d_out, nullptr, DMODEL, nullptr, x1, FFDIM);
}
